// FAVORCirculantSTRINGAttention_34205119545595
// MI455X (gfx1250) — compile-verified, matrix operand order checked by reading, not run on hardware
//
#include <hip/hip_runtime.h>
#include <hip/hip_bf16.h>

// ---------------------------------------------------------------------------
// Shapes (fixed by the reference)
// ---------------------------------------------------------------------------
#define BB 32
#define NN 1025
#define CC 768
#define HH 12
#define DD 64
#define MM 256          // 4*D random features
#define ROWS_QKV (BB * NN)          // 32800
#define ROWS_BHN (BB * HH * NN)     // 393600

typedef __attribute__((ext_vector_type(16))) __bf16 v16bf;
typedef __attribute__((ext_vector_type(8)))  float  v8f;

union V16 {
    v16bf v;
    unsigned short u[16];
};

__device__ __forceinline__ unsigned short f2bf(float f) {
    unsigned int x = __float_as_uint(f);
    x += 0x7fffu + ((x >> 16) & 1u);          // round-to-nearest-even
    return (unsigned short)(x >> 16);
}
__device__ __forceinline__ float bf2f(unsigned short s) {
    return __uint_as_float(((unsigned int)s) << 16);
}

// pack 4 floats -> 4 bf16 (8 bytes) into LDS
__device__ __forceinline__ void store_bf4(unsigned short* dst, float a, float b,
                                          float c, float d) {
    uint2 p;
    p.x = (unsigned int)f2bf(a) | ((unsigned int)f2bf(b) << 16);
    p.y = (unsigned int)f2bf(c) | ((unsigned int)f2bf(d) << 16);
    *(uint2*)dst = p;
}

// ---------------------------------------------------------------------------
// gfx1250 async global->LDS copy (ASYNCcnt path); falls back to a sync copy.
// Builtin prototype (from hipcc diagnostic): pointee is int vector_size(16).
// ---------------------------------------------------------------------------
#if __has_builtin(__builtin_amdgcn_global_load_async_to_lds_b128) && \
    __has_builtin(__builtin_amdgcn_s_wait_asynccnt)
#define USE_ASYNC_LDS 1
#else
#define USE_ASYNC_LDS 0
#endif

typedef int b128v __attribute__((vector_size(16)));

__device__ __forceinline__ void g2l_b128(const void* g, void* l) {
#if USE_ASYNC_LDS
    __builtin_amdgcn_global_load_async_to_lds_b128(
        (__attribute__((address_space(1))) b128v*)(unsigned long long)g,
        (__attribute__((address_space(3))) b128v*)(unsigned long long)l,
        0, 0);
#else
    uint4 t = *(const uint4*)g;
    *(uint4*)l = t;
#endif
}
__device__ __forceinline__ void wait_async_lds() {
#if USE_ASYNC_LDS
    __builtin_amdgcn_s_wait_asynccnt(0);
#endif
}

// K-pair index inside a 16-bit 16x32 WMMA A/B fragment (ISA 7.12.2):
// VGPR v<4 : K = 2v + 8*half ; VGPR v>=4 : K = 16 + 2(v-4) + 8*half
// (fills u[] with two contiguous 16B LDS chunks -> compiler emits ds_load_b128)
__device__ __forceinline__ int kmap(int v, int half) {
    return (v < 4) ? (2 * v + 8 * half) : (16 + 2 * (v - 4) + 8 * half);
}

// ---------------------------------------------------------------------------
// Kernel 0: build circulant rotation kernels r[h,p,0:64] from c_row/c_col.
// theta_k = 2*Im(FFT(g))[k] = -2 * sum_d g[d] sin(2*pi*k*d/64)
// r[j]    = (1/64) * sum_k cos(theta_k + 2*pi*k*j/64)     (real by symmetry)
// ---------------------------------------------------------------------------
__global__ __launch_bounds__(64) void build_rot_kernel(
    const float* __restrict__ c_row, const float* __restrict__ c_col,
    float* __restrict__ rker)
{
    __shared__ float theta[64];
    const int hp = blockIdx.x;           // h*1024 + p
    const int h  = hp >> 10;
    const int p  = hp & 1023;
    const float pr = (float)(p >> 5);
    const float pc = (float)(p & 31);
    const int t = threadIdx.x;
    const float w0 = 6.2831853071795864769f / 64.0f;

    float s = 0.f;
    for (int d = 0; d < 64; ++d) {
        float g = pr * c_row[h * 64 + d] + pc * c_col[h * 64 + d];
        s += g * __sinf(w0 * (float)(t * d));
    }
    theta[t] = -2.0f * s;
    __syncthreads();

    float acc = 0.f;
    for (int k = 0; k < 64; ++k)
        acc += __cosf(theta[k] + w0 * (float)(k * t));
    rker[(size_t)hp * 64 + t] = acc * (1.0f / 64.0f);
}

// ---------------------------------------------------------------------------
// Kernel 1 (generic): bf16 WMMA GEMM   OUT[m,n] = sum_k A[m,k]*W[n,k]
//   block tile 128x64, K-step 32, 8 waves, each wave 32x32 (2x2 WMMA)
//   mode 0: f32 out   mode 1: f32 out + bias[n]
//   mode 2: bf16 out = exp(val - rowsub[m])   (FAVOR feature map)
//   Requires: K % 32 == 0, Ncols % 64 == 0 (true for all uses here).
// ---------------------------------------------------------------------------
#define GT_M 128
#define GT_N 64
#define GT_K 32

__global__ __launch_bounds__(256) void gemm_wmma_kernel(
    const float* __restrict__ A, const float* __restrict__ W,
    const float* __restrict__ bias, const float* __restrict__ rowsub,
    float ascale, int Mrows, int K, int Ncols,
    float* __restrict__ outF, unsigned short* __restrict__ outB, int mode)
{
    __shared__ __align__(16) unsigned short sA[GT_M * GT_K];   // [m][k] bf16
    __shared__ __align__(16) unsigned short sW[GT_N * GT_K];   // [n][k] bf16

    const int tid  = threadIdx.x;
    const int m0   = blockIdx.x * GT_M;
    const int n0   = blockIdx.y * GT_N;
    const int wid  = tid >> 5;
    const int lane = tid & 31;
    const int l16  = lane & 15;
    const int half = lane >> 4;
    const int wm   = (wid & 3) * 32;
    const int wn   = (wid >> 2) * 32;
    const bool full = (m0 + GT_M <= Mrows);

    v8f acc[2][2];
    const v8f vz = {0.f,0.f,0.f,0.f,0.f,0.f,0.f,0.f};
    for (int i = 0; i < 2; ++i)
        for (int j = 0; j < 2; ++j) acc[i][j] = vz;

    for (int k0 = 0; k0 < K; k0 += GT_K) {
        // ---- stage A (128x32 f32) : batched float4 loads, then convert ----
        float4 ta[4];
        if (full) {
            #pragma unroll
            for (int i = 0; i < 4; ++i) {
                int c = tid + (i << 8);
                int m = c >> 3, kc = (c & 7) << 2;
                ta[i] = *(const float4*)(A + (size_t)(m0 + m) * K + k0 + kc);
            }
        } else {
            #pragma unroll
            for (int i = 0; i < 4; ++i) {
                int c = tid + (i << 8);
                int m = c >> 3, kc = (c & 7) << 2;
                ta[i] = make_float4(0.f, 0.f, 0.f, 0.f);
                if (m0 + m < Mrows)
                    ta[i] = *(const float4*)(A + (size_t)(m0 + m) * K + k0 + kc);
            }
        }
        // ---- stage W (64x32 f32) : always full ----
        float4 tw[2];
        #pragma unroll
        for (int i = 0; i < 2; ++i) {
            int c = tid + (i << 8);
            int n = c >> 3, kc = (c & 7) << 2;
            tw[i] = *(const float4*)(W + (size_t)(n0 + n) * K + k0 + kc);
        }
        // L2 prefetch of next A tile (gfx1250 global_prefetch)
        if (full && k0 + GT_K < K) {
            int m = tid >> 3, kc = (tid & 7) << 2;
            __builtin_prefetch(A + (size_t)(m0 + m) * K + k0 + GT_K + kc, 0, 1);
        }
        #pragma unroll
        for (int i = 0; i < 4; ++i) {
            int c = tid + (i << 8);
            int m = c >> 3, kc = (c & 7) << 2;
            store_bf4(&sA[m * GT_K + kc], ta[i].x * ascale, ta[i].y * ascale,
                      ta[i].z * ascale, ta[i].w * ascale);
        }
        #pragma unroll
        for (int i = 0; i < 2; ++i) {
            int c = tid + (i << 8);
            int n = c >> 3, kc = (c & 7) << 2;
            store_bf4(&sW[n * GT_K + kc], tw[i].x, tw[i].y, tw[i].z, tw[i].w);
        }
        __syncthreads();

        V16 bfrag[2];
        #pragma unroll
        for (int ni = 0; ni < 2; ++ni) {
            int n = wn + ni * 16 + l16;
            #pragma unroll
            for (int v = 0; v < 8; ++v) {
                int kk = kmap(v, half);
                bfrag[ni].u[2 * v]     = sW[n * GT_K + kk];
                bfrag[ni].u[2 * v + 1] = sW[n * GT_K + kk + 1];
            }
        }
        #pragma unroll
        for (int mi = 0; mi < 2; ++mi) {
            V16 afrag;
            int m = wm + mi * 16 + l16;
            #pragma unroll
            for (int v = 0; v < 8; ++v) {
                int kk = kmap(v, half);
                afrag.u[2 * v]     = sA[m * GT_K + kk];
                afrag.u[2 * v + 1] = sA[m * GT_K + kk + 1];
            }
            #pragma unroll
            for (int ni = 0; ni < 2; ++ni) {
                acc[mi][ni] = __builtin_amdgcn_wmma_f32_16x16x32_bf16(
                    false, afrag.v, false, bfrag[ni].v,
                    (short)0, acc[mi][ni], false, false);
            }
        }
        __syncthreads();
    }

    // epilogue: C/D layout -> row = tile + r + 8*half, col = tile + lane%16
    #pragma unroll
    for (int mi = 0; mi < 2; ++mi) {
        #pragma unroll
        for (int ni = 0; ni < 2; ++ni) {
            #pragma unroll
            for (int r = 0; r < 8; ++r) {
                int row = m0 + wm + mi * 16 + r + 8 * half;
                int col = n0 + wn + ni * 16 + l16;
                if (row < Mrows) {
                    float v = acc[mi][ni][r];
                    if (mode == 2) {
                        v = __expf(v - rowsub[row]);
                        outB[(size_t)row * Ncols + col] = f2bf(v);
                    } else {
                        if (mode == 1) v += bias[col];
                        outF[(size_t)row * Ncols + col] = v;
                    }
                }
            }
        }
    }
}

// ---------------------------------------------------------------------------
// Kernel 2: apply circulant rotation to q,k (token 0 = CLS pass-through),
// scatter to (B,H,N,D) layout, and emit rowsub = 0.5*||x * D^-1/4||^2.
// ---------------------------------------------------------------------------
__global__ __launch_bounds__(64) void rotate_qk_kernel(
    const float* __restrict__ qkv, const float* __restrict__ rker,
    float* __restrict__ q_rot, float* __restrict__ k_rot,
    float* __restrict__ qn, float* __restrict__ kn)
{
    __shared__ float rsh[64];
    __shared__ float xsh[64];
    __shared__ float red[64];
    const int n = blockIdx.x;       // token 0..1024
    const int h = blockIdx.y;       // head
    const int d = threadIdx.x;
    const bool cls = (n == 0);
    if (!cls) {
        int p = n - 1;
        rsh[d] = rker[((size_t)(h << 10) + p) * 64 + d];
    }
    __syncthreads();

    const float nsc = 0.0625f;      // 0.5 * (64^-0.25)^2

    for (int b = 0; b < BB; ++b) {
        for (int s = 0; s < 2; ++s) {   // 0 = q, 1 = k
            const size_t src = ((size_t)(b * NN + n) * 3 + s) * CC + h * DD;
            xsh[d] = qkv[src + d];
            __syncthreads();
            float y;
            if (cls) {
                y = xsh[d];
            } else {
                float a = 0.f;
                #pragma unroll 8
                for (int j = 0; j < 64; ++j)
                    a += rsh[(d - j) & 63] * xsh[j];
                y = a;
            }
            float* dst = s ? k_rot : q_rot;
            const size_t o = ((size_t)((b * HH + h) * NN + n)) * DD + d;
            dst[o] = y;
            red[d] = y * y;
            __syncthreads();
            for (int off = 32; off > 0; off >>= 1) {
                if (d < off) red[d] += red[d + off];
                __syncthreads();
            }
            if (d == 0) {
                float* np = s ? kn : qn;
                np[(size_t)(b * HH + h) * NN + n] = red[0] * nsc;
            }
            __syncthreads();
        }
    }
}

// ---------------------------------------------------------------------------
// Kernel 3: per (b,h)  context[m,e] = sum_n k_feat[n,m] * v[n,e]  (M=256,E=64)
// plus fused k_sum[m]. k_feat tiles (already bf16) go LDS via async DMA.
// ---------------------------------------------------------------------------
__global__ __launch_bounds__(256) void context_kernel(
    const unsigned short* __restrict__ k_feat, const float* __restrict__ qkv,
    float* __restrict__ ctx, float* __restrict__ k_sum)
{
    __shared__ __align__(16) unsigned short sK[32 * 256];   // [n][m] bf16
    __shared__ __align__(16) unsigned short sV[32 * 64];    // [n][e] bf16
    const int bh = blockIdx.x;
    const int b = bh / HH, h = bh % HH;
    const int tid = threadIdx.x;
    const int wid = tid >> 5, lane = tid & 31;
    const int l16 = lane & 15, half = lane >> 4;

    v8f acc[2][4];
    const v8f vz = {0.f,0.f,0.f,0.f,0.f,0.f,0.f,0.f};
    for (int i = 0; i < 2; ++i)
        for (int j = 0; j < 4; ++j) acc[i][j] = vz;
    float ksum = 0.f;

    for (int n0 = 0; n0 < NN; n0 += 32) {
        // k_feat tile 32x256 bf16 : async global->LDS (b128 chunks)
        #pragma unroll
        for (int i = 0; i < 4; ++i) {
            int c = tid + (i << 8);
            int nn = c >> 5, mc = (c & 31) << 3;
            int n = n0 + nn;
            unsigned short* l = &sK[nn * 256 + mc];
            if (n < NN) {
                g2l_b128(&k_feat[((size_t)bh * NN + n) * MM + mc], l);
            } else {
                uint4 z = make_uint4(0, 0, 0, 0);
                *(uint4*)l = z;
            }
        }
        // v tile 32x64 f32 -> bf16 (batched float4 loads)
        float4 tv[2];
        #pragma unroll
        for (int i = 0; i < 2; ++i) {
            int c = tid + (i << 8);
            int nn = c >> 4, ec = (c & 15) << 2;
            int n = n0 + nn;
            tv[i] = make_float4(0.f, 0.f, 0.f, 0.f);
            if (n < NN)
                tv[i] = *(const float4*)&qkv[((size_t)(b * NN + n) * 3 + 2) * CC +
                                             h * DD + ec];
        }
        #pragma unroll
        for (int i = 0; i < 2; ++i) {
            int c = tid + (i << 8);
            int nn = c >> 4, ec = (c & 15) << 2;
            store_bf4(&sV[nn * 64 + ec], tv[i].x, tv[i].y, tv[i].z, tv[i].w);
        }
        wait_async_lds();
        __syncthreads();

        // fused k_sum: thread tid owns feature m = tid
        #pragma unroll 8
        for (int nn = 0; nn < 32; ++nn) ksum += bf2f(sK[nn * 256 + tid]);

        V16 bfrag[4];
        #pragma unroll
        for (int ei = 0; ei < 4; ++ei) {
            int e = ei * 16 + l16;
            #pragma unroll
            for (int v = 0; v < 8; ++v) {
                int kk = kmap(v, half);
                bfrag[ei].u[2 * v]     = sV[kk * 64 + e];
                bfrag[ei].u[2 * v + 1] = sV[(kk + 1) * 64 + e];
            }
        }
        #pragma unroll
        for (int mi = 0; mi < 2; ++mi) {
            int mbase = (wid * 2 + mi) * 16;   // 16 m-tiles over 8 waves
            V16 afrag;                          // A[m, k=n] = k_feat^T (LDS transpose read)
            #pragma unroll
            for (int v = 0; v < 8; ++v) {
                int kk = kmap(v, half);
                afrag.u[2 * v]     = sK[kk * 256 + mbase + l16];
                afrag.u[2 * v + 1] = sK[(kk + 1) * 256 + mbase + l16];
            }
            #pragma unroll
            for (int ei = 0; ei < 4; ++ei) {
                acc[mi][ei] = __builtin_amdgcn_wmma_f32_16x16x32_bf16(
                    false, afrag.v, false, bfrag[ei].v,
                    (short)0, acc[mi][ei], false, false);
            }
        }
        __syncthreads();
    }

    #pragma unroll
    for (int mi = 0; mi < 2; ++mi) {
        #pragma unroll
        for (int ei = 0; ei < 4; ++ei) {
            #pragma unroll
            for (int r = 0; r < 8; ++r) {
                int m = (wid * 2 + mi) * 16 + r + 8 * half;
                int e = ei * 16 + l16;
                ctx[((size_t)bh * MM + m) * DD + e] = acc[mi][ei][r];
            }
        }
    }
    k_sum[(size_t)bh * MM + tid] = ksum;
}

// ---------------------------------------------------------------------------
// Kernel 4: per (b,h, n-tile of 128):
//   num[n,e] = sum_m q_feat[n,m]*ctx[m,e] ; den[n] = sum_m q_feat[n,m]*k_sum[m]
//   attn[b,n,h*64+e] = num / max(den, EPS)
// q_feat tiles (bf16) -> LDS via async DMA.
// ---------------------------------------------------------------------------
__global__ __launch_bounds__(256) void numden_kernel(
    const unsigned short* __restrict__ q_feat, const float* __restrict__ ctx,
    const float* __restrict__ k_sum, float* __restrict__ attn)
{
    __shared__ __align__(16) unsigned short sQ[128 * 32];   // [n][m] bf16
    __shared__ __align__(16) unsigned short sC[32 * 64];    // [m][e] bf16
    __shared__ float sKS[32];
    __shared__ float sDen[128];
    const int nbase = blockIdx.x * 128;
    const int bh = blockIdx.y;
    const int b = bh / HH, h = bh % HH;
    const int tid = threadIdx.x;
    const int wid = tid >> 5, lane = tid & 31;
    const int l16 = lane & 15, half = lane >> 4;

    v8f acc[4];
    const v8f vz = {0.f,0.f,0.f,0.f,0.f,0.f,0.f,0.f};
    for (int i = 0; i < 4; ++i) acc[i] = vz;
    float den = 0.f;

    for (int m0 = 0; m0 < MM; m0 += 32) {
        // q_feat tile 128x32 bf16 : async global->LDS (b128 chunks)
        #pragma unroll
        for (int i = 0; i < 2; ++i) {
            int c = tid + (i << 8);
            int nl = c >> 2, mc = (c & 3) << 3;
            int n = nbase + nl;
            unsigned short* l = &sQ[nl * 32 + mc];
            if (n < NN) {
                g2l_b128(&q_feat[((size_t)bh * NN + n) * MM + m0 + mc], l);
            } else {
                uint4 z = make_uint4(0, 0, 0, 0);
                *(uint4*)l = z;
            }
        }
        // ctx tile 32x64 f32 -> bf16 (no guards needed)
        float4 tc[2];
        #pragma unroll
        for (int i = 0; i < 2; ++i) {
            int c = tid + (i << 8);
            int ml = c >> 4, ec = (c & 15) << 2;
            tc[i] = *(const float4*)&ctx[((size_t)bh * MM + m0 + ml) * DD + ec];
        }
        #pragma unroll
        for (int i = 0; i < 2; ++i) {
            int c = tid + (i << 8);
            int ml = c >> 4, ec = (c & 15) << 2;
            store_bf4(&sC[ml * 64 + ec], tc[i].x, tc[i].y, tc[i].z, tc[i].w);
        }
        if (tid < 32) sKS[tid] = k_sum[(size_t)bh * MM + m0 + tid];
        wait_async_lds();
        __syncthreads();

        if (tid < 128) {
            #pragma unroll 8
            for (int j = 0; j < 32; ++j)
                den += bf2f(sQ[tid * 32 + j]) * sKS[j];
        }

        V16 afrag;
        int nr = wid * 16 + l16;              // 8 n-strips of 16 over 8 waves
        #pragma unroll
        for (int v = 0; v < 8; ++v) {
            int kk = kmap(v, half);
            afrag.u[2 * v]     = sQ[nr * 32 + kk];
            afrag.u[2 * v + 1] = sQ[nr * 32 + kk + 1];
        }
        #pragma unroll
        for (int ei = 0; ei < 4; ++ei) {
            V16 bfrag;
            #pragma unroll
            for (int v = 0; v < 8; ++v) {
                int kk = kmap(v, half);
                bfrag.u[2 * v]     = sC[kk * 64 + ei * 16 + l16];
                bfrag.u[2 * v + 1] = sC[(kk + 1) * 64 + ei * 16 + l16];
            }
            acc[ei] = __builtin_amdgcn_wmma_f32_16x16x32_bf16(
                false, afrag.v, false, bfrag.v, (short)0, acc[ei], false, false);
        }
        __syncthreads();
    }

    if (tid < 128) sDen[tid] = fmaxf(den, 1e-6f);
    __syncthreads();

    #pragma unroll
    for (int ei = 0; ei < 4; ++ei) {
        #pragma unroll
        for (int r = 0; r < 8; ++r) {
            int nl = wid * 16 + r + 8 * half;
            int n = nbase + nl;
            if (n < NN) {
                float v = acc[ei][r] / sDen[nl];
                attn[((size_t)(b * NN + n)) * CC + h * DD + ei * 16 + l16] = v;
            }
        }
    }
}

// ---------------------------------------------------------------------------
// Host-side launcher
// ---------------------------------------------------------------------------
extern "C" void kernel_launch(void* const* d_in, const int* in_sizes, int n_in,
                              void* d_out, int out_size, void* d_ws, size_t ws_size,
                              hipStream_t stream)
{
    const float* x         = (const float*)d_in[0];   // (B,N,C)
    const float* w_qkv     = (const float*)d_in[1];   // (3C,C)
    const float* w_proj    = (const float*)d_in[2];   // (C,C)
    const float* b_proj    = (const float*)d_in[3];   // (C,)
    const float* c_row     = (const float*)d_in[4];   // (H,D)
    const float* c_col     = (const float*)d_in[5];   // (H,D)
    const float* proj_feat = (const float*)d_in[6];   // (M,D)
    float* out = (float*)d_out;                       // (B,N,C) f32

    char* ws = (char*)d_ws;
    size_t off = 0;
    auto carve = [&](size_t bytes) {
        char* p = ws + off;
        off = (off + bytes + 255) & ~(size_t)255;
        return p;
    };
    float* qkv   = (float*)carve((size_t)ROWS_QKV * 3 * CC * 4);   // 302 MB
    float* rk    = (float*)carve((size_t)HH * 1024 * DD * 4);      //   3 MB
    float* q_rot = (float*)carve((size_t)ROWS_BHN * DD * 4);       // 101 MB
    float* k_rot = (float*)carve((size_t)ROWS_BHN * DD * 4);       // 101 MB
    float* qn    = (float*)carve((size_t)ROWS_BHN * 4);            // 1.6 MB
    float* kn    = (float*)carve((size_t)ROWS_BHN * 4);            // 1.6 MB
    float* ctx   = (float*)carve((size_t)BB * HH * MM * DD * 4);   //  25 MB
    float* ksum  = (float*)carve((size_t)BB * HH * MM * 4);        // 0.4 MB
    float* attn  = (float*)carve((size_t)ROWS_QKV * CC * 4);       // 101 MB
    unsigned short* qf = (unsigned short*)carve((size_t)ROWS_BHN * MM * 2); // 202 MB
    unsigned short* kf = (unsigned short*)carve((size_t)ROWS_BHN * MM * 2); // 202 MB

    const float XS_SCALE = 0.35355339059327373f;      // 64^-0.25

    build_rot_kernel<<<HH * 1024, 64, 0, stream>>>(c_row, c_col, rk);

    gemm_wmma_kernel<<<dim3((ROWS_QKV + GT_M - 1) / GT_M, (3 * CC) / GT_N), 256, 0, stream>>>(
        x, w_qkv, nullptr, nullptr, 1.0f, ROWS_QKV, CC, 3 * CC, qkv, nullptr, 0);

    rotate_qk_kernel<<<dim3(NN, HH), 64, 0, stream>>>(qkv, rk, q_rot, k_rot, qn, kn);

    gemm_wmma_kernel<<<dim3(ROWS_BHN / GT_M, MM / GT_N), 256, 0, stream>>>(
        q_rot, proj_feat, nullptr, qn, XS_SCALE, ROWS_BHN, DD, MM, nullptr, qf, 2);
    gemm_wmma_kernel<<<dim3(ROWS_BHN / GT_M, MM / GT_N), 256, 0, stream>>>(
        k_rot, proj_feat, nullptr, kn, XS_SCALE, ROWS_BHN, DD, MM, nullptr, kf, 2);

    context_kernel<<<BB * HH, 256, 0, stream>>>(kf, qkv, ctx, ksum);

    numden_kernel<<<dim3((NN + 127) / 128, BB * HH), 256, 0, stream>>>(qf, ctx, ksum, attn);

    gemm_wmma_kernel<<<dim3((ROWS_QKV + GT_M - 1) / GT_M, CC / GT_N), 256, 0, stream>>>(
        attn, w_proj, b_proj, nullptr, 1.0f, ROWS_QKV, CC, CC, out, nullptr, 1);
}
